// SRNLayer_42872363549117
// MI455X (gfx1250) — matrix operand order repaired and checked by reading, not compiled, not run on hardware
//
#include <hip/hip_runtime.h>

// ---------------------------------------------------------------------------
// Problem constants (match reference)
// ---------------------------------------------------------------------------
#define NB     4
#define NSEQ   4096
#define DIM    1024
#define NH     8
#define DH     128
#define KSLOT  64
#define WINSZ  32
#define DCC    256
#define T_TOK  (NB * NSEQ)          // 16384 tokens
#define ATT_SCALE 0.088388347648318447f   // 1/sqrt(128)

typedef __bf16 bf16_t;
typedef __attribute__((ext_vector_type(16))) __bf16 v16bf;
typedef __attribute__((ext_vector_type(8)))  float  v8f;
typedef __attribute__((ext_vector_type(4)))  int    v4i;

static __device__ inline v8f zero8() {
    v8f z;
    for (int i = 0; i < 8; ++i) z[i] = 0.0f;
    return z;
}

// ---------------------------------------------------------------------------
// gfx1250 async global->LDS DMA (ASYNCcnt path), guarded so the file compiles
// on toolchains (and the host pass) that lack the builtin.
// Signature (from clang diagnostic): (v4i AS1*, v4i AS3*, imm int, imm int).
// ---------------------------------------------------------------------------
#if defined(__AMDGCN__) && __has_builtin(__builtin_amdgcn_global_load_async_to_lds_b128)
#define HAVE_ASYNC_LDS 1
static __device__ inline void async_copy16(const bf16_t* g, bf16_t* l) {
    __builtin_amdgcn_global_load_async_to_lds_b128(
        (__attribute__((address_space(1))) v4i*)(uintptr_t)(const void*)g,
        (__attribute__((address_space(3))) v4i*)(unsigned)(uintptr_t)(void*)l,
        0, 0);
}
#else
#define HAVE_ASYNC_LDS 0
#endif

// ---------------------------------------------------------------------------
// fp32 -> bf16 conversion (plain, for A-side weights)
// ---------------------------------------------------------------------------
__global__ __launch_bounds__(256)
void f32_to_bf16_kernel(const float* __restrict__ src, bf16_t* __restrict__ dst, int n) {
    int i = blockIdx.x * 256 + threadIdx.x;
    if (i < n) dst[i] = (bf16_t)src[i];
}

// ---------------------------------------------------------------------------
// fp32 [Krows x Ncols] -> bf16 transposed [Ncols x Krows] (B-side weights)
// write-coalesced; weights are small so the strided reads are fine.
// ---------------------------------------------------------------------------
__global__ __launch_bounds__(256)
void f32_to_bf16_transpose_kernel(const float* __restrict__ src, bf16_t* __restrict__ dst,
                                  int Krows, int Ncols) {
    int i = blockIdx.x * 256 + threadIdx.x;
    if (i >= Krows * Ncols) return;
    int n = i / Krows;
    int k = i - n * Krows;
    dst[(size_t)n * Krows + k] = (bf16_t)src[(size_t)k * Ncols + n];
}

// ---------------------------------------------------------------------------
// LayerNorm: one block (256 thr) per row of D=1024, writes bf16
// ---------------------------------------------------------------------------
__global__ __launch_bounds__(256)
void ln_bf16_kernel(const float* __restrict__ x, const float* __restrict__ g,
                    const float* __restrict__ b, bf16_t* __restrict__ out) {
    __shared__ float red[256];
    const int row = blockIdx.x;
    const int tid = threadIdx.x;

    const float4 v = ((const float4*)(x + (size_t)row * DIM))[tid];
    red[tid] = v.x + v.y + v.z + v.w;
    __syncthreads();
    for (int o = 128; o > 0; o >>= 1) { if (tid < o) red[tid] += red[tid + o]; __syncthreads(); }
    const float mean = red[0] * (1.0f / DIM);
    __syncthreads();

    const float dx = v.x - mean, dy = v.y - mean, dz = v.z - mean, dw = v.w - mean;
    red[tid] = dx * dx + dy * dy + dz * dz + dw * dw;
    __syncthreads();
    for (int o = 128; o > 0; o >>= 1) { if (tid < o) red[tid] += red[tid + o]; __syncthreads(); }
    const float rstd = rsqrtf(red[0] * (1.0f / DIM) + 1e-5f);

    const float4 gv = ((const float4*)g)[tid];
    const float4 bv = ((const float4*)b)[tid];
    union { bf16_t h[4]; uint2 q; } po;
    po.h[0] = (bf16_t)(dx * rstd * gv.x + bv.x);
    po.h[1] = (bf16_t)(dy * rstd * gv.y + bv.y);
    po.h[2] = (bf16_t)(dz * rstd * gv.z + bv.z);
    po.h[3] = (bf16_t)(dw * rstd * gv.w + bv.w);
    *(uint2*)&out[(size_t)row * DIM + tid * 4] = po.q;
}

// ---------------------------------------------------------------------------
// Causal windowed mean over sequence, per (batch, dim) lane. bf16 in/out.
// ---------------------------------------------------------------------------
__global__ __launch_bounds__(256)
void winmean_kernel(const bf16_t* __restrict__ h, bf16_t* __restrict__ xw) {
    const int id = blockIdx.x * 256 + threadIdx.x;   // 0 .. NB*DIM-1
    if (id >= NB * DIM) return;
    const int b = id / DIM;
    const int d = id % DIM;
    const bf16_t* src = h  + (size_t)b * NSEQ * DIM + d;
    bf16_t*       dst = xw + (size_t)b * NSEQ * DIM + d;
    float s = 0.0f;
    for (int n = 0; n < NSEQ; ++n) {
        s += (float)src[(size_t)n * DIM];
        if (n >= WINSZ) s -= (float)src[(size_t)(n - WINSZ) * DIM];
        const float cnt = (float)((n + 1) < WINSZ ? (n + 1) : WINSZ);
        dst[(size_t)n * DIM] = (bf16_t)(s / cnt);
    }
}

// ---------------------------------------------------------------------------
// Slot attention: block per token, wave per head.
// ---------------------------------------------------------------------------
__global__ __launch_bounds__(256)
void slot_attention_kernel(const float* __restrict__ Q,
                           const float* __restrict__ slot_keys,
                           const float* __restrict__ gate,
                           const float* __restrict__ pos_bias,
                           const float* __restrict__ Vs,
                           bf16_t* __restrict__ routed) {
    __shared__ float sQ[DIM];
    __shared__ float sW[NH][KSLOT];

    const int t   = blockIdx.x;
    const int n   = t % NSEQ;
    const int tid = threadIdx.x;
    const int h    = tid >> 5;
    const int lane = tid & 31;

    ((float4*)sQ)[tid] = ((const float4*)(Q + (size_t)t * DIM))[tid];
    __syncthreads();

    const float* qh = sQ + h * DH;
    float sc[2];
    #pragma unroll
    for (int sidx = 0; sidx < 2; ++sidx) {
        const int k = lane + sidx * 32;
        const float* kr = slot_keys + (size_t)k * DIM + h * DH;
        float d = 0.0f;
        #pragma unroll 4
        for (int j = 0; j < DH; ++j) d += qh[j] * kr[j];
        const float g = gate[(size_t)t * KSLOT + k];
        sc[sidx] = d * ATT_SCALE * g + pos_bias[(size_t)n * KSLOT + k];
    }
    float mx = fmaxf(sc[0], sc[1]);
    for (int o = 16; o > 0; o >>= 1) mx = fmaxf(mx, __shfl_xor(mx, o, 32));
    const float e0 = __expf(sc[0] - mx);
    const float e1 = __expf(sc[1] - mx);
    float sum = e0 + e1;
    for (int o = 16; o > 0; o >>= 1) sum += __shfl_xor(sum, o, 32);
    const float inv = 1.0f / sum;
    sW[h][lane]      = e0 * inv;
    sW[h][lane + 32] = e1 * inv;
    __syncthreads();

    float acc0 = 0.f, acc1 = 0.f, acc2 = 0.f, acc3 = 0.f;
    const int d0 = lane * 4;
    for (int k = 0; k < KSLOT; ++k) {
        const float w = sW[h][k];
        const float4 v = *(const float4*)(Vs + (size_t)k * DIM + h * DH + d0);
        acc0 += w * v.x; acc1 += w * v.y; acc2 += w * v.z; acc3 += w * v.w;
    }
    union { bf16_t bh[4]; uint2 q; } po;
    po.bh[0] = (bf16_t)acc0; po.bh[1] = (bf16_t)acc1;
    po.bh[2] = (bf16_t)acc2; po.bh[3] = (bf16_t)acc3;
    *(uint2*)&routed[(size_t)t * DIM + h * DH + d0] = po.q;
}

// ---------------------------------------------------------------------------
// Generic bf16 WMMA GEMM: C = A[M x K] * B[K x N] with B supplied TRANSPOSED
// (BT is [N x K], row-major). 128x128 block tile, BK=32, 8 waves (2x4), each
// wave computes 64x32 via 4x2 v_wmma_f32_16x16x32_bf16 tiles.
// Split-K A source: global k < kSplit reads A (lda), else A2 (lda2).
// Async double-buffered global->LDS DMA when the gfx1250 builtin exists.
// MODE: 0 = outF = acc+bias
//       2 = outB = gelu(acc+bias)                       (erf gelu, bf16)
//       3 = outF = res + acc+bias                       (residual)
//       4 = outF = max(sigmoid(acc+bias), 1e-4)         (gate)
//       5 = outF = res + sigmoid(acc+bias) * expb       (final fuse)
// ---------------------------------------------------------------------------
#define BM 128
#define BN 128
#define BKK 32
#define LSTR 48   // padded LDS row stride in bf16 elems (96 B, 16B aligned)
#if HAVE_ASYNC_LDS
#define NBUF 2
#else
#define NBUF 1
#endif

template <int MODE>
__global__ __launch_bounds__(256)
void gemm_wmma_bf16(const bf16_t* __restrict__ A,  int lda,
                    const bf16_t* __restrict__ A2, int lda2, int kSplit,
                    const bf16_t* __restrict__ BT,
                    const float* __restrict__ bias,
                    int M, int Nn, int K,
                    float* __restrict__ outF, bf16_t* __restrict__ outB,
                    const float* __restrict__ res, const float* __restrict__ expb) {
    __shared__ __align__(16) bf16_t As[NBUF][BM * LSTR];
    __shared__ __align__(16) bf16_t Bs[NBUF][BN * LSTR];

    const int tid    = threadIdx.x;
    const int lane   = tid & 31;
    const int wave   = tid >> 5;
    const int wr     = wave >> 2;       // 0..1   (M direction, 64 rows each)
    const int wc     = wave & 3;        // 0..3   (N direction, 32 cols each)
    const int lanelo = lane & 15;
    const int hi     = lane >> 4;

    const int blockM = blockIdx.y * BM;
    const int blockN = blockIdx.x * BN;

    union AF { v16bf v; unsigned u[8]; };
    v8f acc[4][2];
    for (int m = 0; m < 4; ++m)
        for (int nn = 0; nn < 2; ++nn) acc[m][nn] = zero8();

    // Staging mapping: 4 chunk-threads per row/col, chunk = 8 bf16 (16 B);
    // 256 threads cover 64 rows/cols per pass, 2 passes -> 128.
    const int chunk = tid & 3;
    const int r0    = tid >> 2;

    // Branch-free OOB: clamp addresses; garbage only feeds epilogue-skipped
    // rows/cols, which are never stored.
    auto stage = [&](int buf, int k0) {
        const int kg = k0 + chunk * 8;
        #pragma unroll
        for (int rr = 0; rr < 2; ++rr) {
            const int r  = r0 + rr * 64;
            int ga = blockM + r; if (ga >= M)  ga = M - 1;
            int gb = blockN + r; if (gb >= Nn) gb = Nn - 1;
            const bf16_t* asrc = (kg < kSplit)
                ? (A  + (size_t)ga * lda  + kg)
                : (A2 + (size_t)ga * lda2 + (kg - kSplit));
            const bf16_t* bsrc = BT + (size_t)gb * K + kg;
#if HAVE_ASYNC_LDS
            async_copy16(asrc, &As[buf][r * LSTR + chunk * 8]);
            async_copy16(bsrc, &Bs[buf][r * LSTR + chunk * 8]);
#else
            *(uint4*)&As[buf][r * LSTR + chunk * 8] = *(const uint4*)asrc;
            *(uint4*)&Bs[buf][r * LSTR + chunk * 8] = *(const uint4*)bsrc;
#endif
        }
    };

    auto compute = [&](int buf) {
        AF aF[4], bF[2];
        #pragma unroll
        for (int m = 0; m < 4; ++m) {
            const unsigned* ar = (const unsigned*)&As[buf][(wr * 64 + m * 16 + lanelo) * LSTR];
            #pragma unroll
            for (int p = 0; p < 8; ++p) {
                // A 16x32 bf16: lanes 0-15 pairs K={0..7,16..23}, lanes 16-31 +8
                const int idx = (p < 4 ? p : p + 4) + 4 * hi;   // uint index = k/2
                aF[m].u[p] = ar[idx];
            }
        }
        #pragma unroll
        for (int nn = 0; nn < 2; ++nn) {
            const unsigned* br = (const unsigned*)&Bs[buf][(wc * 32 + nn * 16 + lanelo) * LSTR];
            #pragma unroll
            for (int p = 0; p < 8; ++p)
                bF[nn].u[p] = br[(p < 4 ? p : p + 4) + 4 * hi]; // B: same k order as A
        }
        #pragma unroll
        for (int m = 0; m < 4; ++m)
            #pragma unroll
            for (int nn = 0; nn < 2; ++nn)
                acc[m][nn] = __builtin_amdgcn_wmma_f32_16x16x32_bf16(
                    false, aF[m].v, false, bF[nn].v,
                    (short)0, acc[m][nn], false, false);
    };

    const int nt = K / BKK;
#if HAVE_ASYNC_LDS
    int cur = 0;
    stage(0, 0);
    for (int kt = 0; kt < nt; ++kt) {
        asm volatile("s_wait_asynccnt 0x0" ::: "memory");
        __syncthreads();                       // cur tile visible to all waves
        if (kt + 1 < nt) stage(cur ^ 1, (kt + 1) * BKK);  // DMA overlaps WMMAs
        compute(cur);
        __syncthreads();                       // all done reading cur
        cur ^= 1;
    }
#else
    for (int kt = 0; kt < nt; ++kt) {
        stage(0, kt * BKK);
        __syncthreads();
        compute(0);
        __syncthreads();
    }
#endif

    // ---- epilogue ----
    for (int m = 0; m < 4; ++m) {
        for (int nn = 0; nn < 2; ++nn) {
            const int col = blockN + wc * 32 + nn * 16 + lanelo;
            if (col >= Nn) continue;
            const float bv = bias[col];
            const int rbase = blockM + wr * 64 + m * 16 + hi * 8;
            #pragma unroll
            for (int i = 0; i < 8; ++i) {
                const int row = rbase + i;
                if (row >= M) continue;
                const size_t idx = (size_t)row * Nn + col;
                const float v = acc[m][nn][i] + bv;
                if (MODE == 0) {
                    outF[idx] = v;
                } else if (MODE == 2) {
                    outB[idx] = (bf16_t)(0.5f * v * (1.0f + erff(v * 0.70710678118f)));
                } else if (MODE == 3) {
                    outF[idx] = res[idx] + v;
                } else if (MODE == 4) {
                    outF[idx] = fmaxf(1.0f / (1.0f + __expf(-v)), 1e-4f);
                } else if (MODE == 5) {
                    const float s = 1.0f / (1.0f + __expf(-v));
                    outF[idx] = res[idx] + s * expb[idx];
                }
            }
        }
    }
}

// ---------------------------------------------------------------------------
// Host orchestration
// ---------------------------------------------------------------------------
static inline dim3 gemm_grid(int M, int Nn) {
    return dim3((unsigned)((Nn + BN - 1) / BN), (unsigned)((M + BM - 1) / BM));
}

extern "C" void kernel_launch(void* const* d_in, const int* in_sizes, int n_in,
                              void* d_out, int out_size, void* d_ws, size_t ws_size,
                              hipStream_t stream) {
    (void)in_sizes; (void)n_in; (void)out_size; (void)ws_size;

    const float* x        = (const float*)d_in[0];
    const float* ln1_g    = (const float*)d_in[1];
    const float* ln1_b    = (const float*)d_in[2];
    const float* Wq       = (const float*)d_in[3];
    const float* bq       = (const float*)d_in[4];
    const float* slotk    = (const float*)d_in[5];
    const float* Wsv      = (const float*)d_in[6];
    const float* bsv      = (const float*)d_in[7];
    const float* Wgate    = (const float*)d_in[8];
    const float* bgate    = (const float*)d_in[9];
    const float* pos_bias = (const float*)d_in[10];
    const float* Wo       = (const float*)d_in[11];
    const float* bo       = (const float*)d_in[12];
    const float* ln2_g    = (const float*)d_in[13];
    const float* ln2_b    = (const float*)d_in[14];
    const float* Wc       = (const float*)d_in[15];
    const float* bc       = (const float*)d_in[16];
    const float* Wp       = (const float*)d_in[17];
    const float* bp       = (const float*)d_in[18];
    const float* We       = (const float*)d_in[19];
    const float* be       = (const float*)d_in[20];
    const float* Wgp      = (const float*)d_in[21];
    const float* bgp      = (const float*)d_in[22];
    float* out            = (float*)d_out;

    // ---- workspace layout ----
    char* ws = (char*)d_ws;
    size_t off = 0;
    auto alloc = [&](size_t bytes) -> char* {
        off = (off + 255) & ~(size_t)255;
        char* p = ws + off;
        off += bytes;
        return p;
    };

    const size_t szDD  = (size_t)DIM * DIM;        // 1,048,576
    const size_t szDK  = (size_t)DIM * KSLOT;      // 65,536
    const size_t szDC  = (size_t)DIM * DCC;        // 262,144
    const size_t szCC  = (size_t)DCC * DCC;        // 65,536
    const size_t szGP  = (size_t)(DIM + DCC) * DIM;// 1,310,720
    const size_t szKD  = (size_t)KSLOT * DIM;      // 65,536
    const size_t szTD  = (size_t)T_TOK * DIM;      // 16,777,216
    const size_t szTC  = (size_t)T_TOK * DCC;      // 4,194,304
    const size_t szTK  = (size_t)T_TOK * KSLOT;    // 1,048,576

    // B-side weights, stored transposed [N x K] in bf16
    bf16_t* WqT    = (bf16_t*)alloc(szDD * 2);
    bf16_t* WsvT   = (bf16_t*)alloc(szDD * 2);
    bf16_t* WgateT = (bf16_t*)alloc(szDK * 2);
    bf16_t* WoT    = (bf16_t*)alloc(szDD * 2);
    bf16_t* WcT    = (bf16_t*)alloc(szDC * 2);
    bf16_t* WpT    = (bf16_t*)alloc(szCC * 2);
    bf16_t* WeT    = (bf16_t*)alloc(szDC * 2);
    bf16_t* WgpT   = (bf16_t*)alloc(szGP * 2);    // [1024 x 1280]
    bf16_t* slotk_b= (bf16_t*)alloc(szKD * 2);    // A-side, plain layout

    bf16_t* hb       = (bf16_t*)alloc(szTD * 2);  // LN output (h1, later h2)
    float*  Qbuf     = (float*) alloc(szTD * 4);  // Q; reused as `expanded`
    bf16_t* xwbuf    = (bf16_t*)alloc(szTD * 2);  // windowed mean; reused as `routed`
    float*  gatebuf  = (float*) alloc(szTK * 4);
    float*  Vsbuf    = (float*) alloc(szKD * 4);
    float*  x1buf    = (float*) alloc(szTD * 4);  // x after router residual
    bf16_t* compbuf  = (bf16_t*)alloc(szTC * 2);
    bf16_t* procbuf  = (bf16_t*)alloc(szTC * 2);
    float*  expanded = Qbuf;                      // alias (Q dead after attention)
    bf16_t* routed   = xwbuf;                     // alias (xw dead after gate GEMM)

    // ---- weight conversions (B side transposed) ----
    auto convT = [&](const float* s, bf16_t* d, int Krows, int Ncols) {
        const size_t n = (size_t)Krows * Ncols;
        f32_to_bf16_transpose_kernel<<<(unsigned)((n + 255) / 256), 256, 0, stream>>>(
            s, d, Krows, Ncols);
    };
    convT(Wq, WqT, DIM, DIM);
    convT(Wsv, WsvT, DIM, DIM);
    convT(Wgate, WgateT, DIM, KSLOT);
    convT(Wo, WoT, DIM, DIM);
    convT(Wc, WcT, DIM, DCC);
    convT(Wp, WpT, DCC, DCC);
    convT(We, WeT, DCC, DIM);
    convT(Wgp, WgpT, DIM + DCC, DIM);
    f32_to_bf16_kernel<<<(unsigned)((szKD + 255) / 256), 256, 0, stream>>>(
        slotk, slotk_b, (int)szKD);

    // ---- 1) h1 = LN1(x) -> hb (bf16) ----
    ln_bf16_kernel<<<T_TOK, 256, 0, stream>>>(x, ln1_g, ln1_b, hb);

    // ---- 2) Q = h1 @ Wq + bq  (f32 out) ----
    gemm_wmma_bf16<0><<<gemm_grid(T_TOK, DIM), 256, 0, stream>>>(
        hb, DIM, hb, DIM, DIM, WqT, bq, T_TOK, DIM, DIM,
        Qbuf, nullptr, nullptr, nullptr);

    // ---- 3) xw = causal windowed mean of h1 (bf16) ----
    winmean_kernel<<<(NB * DIM + 255) / 256, 256, 0, stream>>>(hb, xwbuf);

    // ---- 4) gate = clamp(sigmoid(xw @ Wgate + bgate)) (f32, T x 64) ----
    gemm_wmma_bf16<4><<<gemm_grid(T_TOK, KSLOT), 256, 0, stream>>>(
        xwbuf, DIM, xwbuf, DIM, DIM, WgateT, bgate, T_TOK, KSLOT, DIM,
        gatebuf, nullptr, nullptr, nullptr);

    // ---- 5) Vs = slot_keys @ Wsv + bsv (f32, 64 x 1024) ----
    gemm_wmma_bf16<0><<<gemm_grid(KSLOT, DIM), 256, 0, stream>>>(
        slotk_b, DIM, slotk_b, DIM, DIM, WsvT, bsv, KSLOT, DIM, DIM,
        Vsbuf, nullptr, nullptr, nullptr);

    // ---- 6) slot attention -> routed (bf16) ----
    slot_attention_kernel<<<T_TOK, 256, 0, stream>>>(
        Qbuf, slotk, gatebuf, pos_bias, Vsbuf, routed);

    // ---- 7) x1 = x + routed @ Wo + bo (f32) ----
    gemm_wmma_bf16<3><<<gemm_grid(T_TOK, DIM), 256, 0, stream>>>(
        routed, DIM, routed, DIM, DIM, WoT, bo, T_TOK, DIM, DIM,
        x1buf, nullptr, x, nullptr);

    // ---- 8) h2 = LN2(x1) -> hb (bf16, reuse) ----
    ln_bf16_kernel<<<T_TOK, 256, 0, stream>>>(x1buf, ln2_g, ln2_b, hb);

    // ---- 9) compressed = gelu(h2 @ Wc + bc) (bf16, T x 256) ----
    gemm_wmma_bf16<2><<<gemm_grid(T_TOK, DCC), 256, 0, stream>>>(
        hb, DIM, hb, DIM, DIM, WcT, bc, T_TOK, DCC, DIM,
        nullptr, compbuf, nullptr, nullptr);

    // ---- 10) processed = gelu(compressed @ Wp + bp) (bf16, T x 256) ----
    gemm_wmma_bf16<2><<<gemm_grid(T_TOK, DCC), 256, 0, stream>>>(
        compbuf, DCC, compbuf, DCC, DCC, WpT, bp, T_TOK, DCC, DCC,
        nullptr, procbuf, nullptr, nullptr);

    // ---- 11) expanded = processed @ We + be (f32, aliases Qbuf) ----
    gemm_wmma_bf16<0><<<gemm_grid(T_TOK, DIM), 256, 0, stream>>>(
        procbuf, DCC, procbuf, DCC, DCC, WeT, be, T_TOK, DIM, DCC,
        expanded, nullptr, nullptr, nullptr);

    // ---- 12) out = x1 + sigmoid([h2|processed] @ Wgp + bgp) * expanded ----
    //      split-K GEMM: K = 1280, k<1024 from h2 (lda=1024), else processed
    gemm_wmma_bf16<5><<<gemm_grid(T_TOK, DIM), 256, 0, stream>>>(
        hb, DIM, procbuf, DCC, DIM, WgpT, bgp, T_TOK, DIM, DIM + DCC,
        out, nullptr, x1buf, expanded);
}
